// MoE_41635412967986
// MI455X (gfx1250) — matrix operand order repaired and checked by reading, not compiled, not run on hardware
//
#include <hip/hip_runtime.h>
#include <hip/hip_bf16.h>

typedef __attribute__((ext_vector_type(16))) __bf16 v16bf;
typedef __attribute__((ext_vector_type(8)))  float  v8f;

#define NTOK   2048      // B*S
#define DDIM   1024
#define FDIM   2048
#define NEXP   8
#define TILE_M 32        // tokens per ffn block
#define NCOLS  256       // N columns staged per pass
#define XA_RS  516       // uints per X row in LDS (512 data + 4 pad -> conflict-free)
#define FU_RS  1028      // uints per fuse row in LDS (1024 data + 4 pad)
#define B_RS   20        // uints per B-stage row (16 data + 4 pad)

__device__ __forceinline__ unsigned short f2bf(float f) {
  unsigned int u = __float_as_uint(f);
  u += 0x7FFFu + ((u >> 16) & 1u);          // round-to-nearest-even
  return (unsigned short)(u >> 16);
}

// ---------------- tiny kernels ----------------

__global__ void zero_cnt(int* cnt) {
  if (threadIdx.x < NEXP) cnt[threadIdx.x] = 0;
}

__global__ void cvt_x(const float* __restrict__ x, unsigned int* __restrict__ xb) {
  int i = blockIdx.x * blockDim.x + threadIdx.x;   // over bf16-pair index
  if (i < NTOK * DDIM / 2) {
    float2 f = ((const float2*)x)[i];
    xb[i] = (unsigned int)f2bf(f.x) | ((unsigned int)f2bf(f.y) << 16);
  }
}

// bulk f32 -> packed bf16 pair conversion (one-time per launch, BW-bound)
__global__ void cvt_w(const float* __restrict__ src, unsigned int* __restrict__ dst,
                      int npairs) {
  int i = blockIdx.x * blockDim.x + threadIdx.x;
  if (i < npairs) {
    float2 f = ((const float2*)src)[i];
    dst[i] = (unsigned int)f2bf(f.x) | ((unsigned int)f2bf(f.y) << 16);
  }
}

// one wave (32 lanes) per token
__global__ void router(const float* __restrict__ x, const float* __restrict__ R,
                       int* cnt, int* ltok, float* lw) {
  int wave = threadIdx.x >> 5, lane = threadIdx.x & 31;
  int t = blockIdx.x * 8 + wave;
  const float* xr = x + (size_t)t * DDIM;
  float acc[NEXP];
#pragma unroll
  for (int e = 0; e < NEXP; ++e) acc[e] = 0.f;
  for (int d = lane; d < DDIM; d += 32) {
    float xv = xr[d];
    const float4* r4 = (const float4*)(R + (size_t)d * NEXP);
    float4 r0 = r4[0], r1 = r4[1];
    acc[0] += xv * r0.x; acc[1] += xv * r0.y; acc[2] += xv * r0.z; acc[3] += xv * r0.w;
    acc[4] += xv * r1.x; acc[5] += xv * r1.y; acc[6] += xv * r1.z; acc[7] += xv * r1.w;
  }
#pragma unroll
  for (int e = 0; e < NEXP; ++e)
#pragma unroll
    for (int off = 16; off > 0; off >>= 1)
      acc[e] += __shfl_down(acc[e], off, 32);
  if (lane == 0) {
    float m = acc[0];
    for (int e = 1; e < NEXP; ++e) m = fmaxf(m, acc[e]);
    float p[NEXP], s = 0.f;
    for (int e = 0; e < NEXP; ++e) { p[e] = __expf(acc[e] - m); s += p[e]; }
    float inv = __builtin_amdgcn_rcpf(s);
    for (int e = 0; e < NEXP; ++e) p[e] *= inv;
    int i0 = 0;
    for (int e = 1; e < NEXP; ++e) if (p[e] > p[i0]) i0 = e;
    int i1 = (i0 == 0) ? 1 : 0;
    for (int e = 0; e < NEXP; ++e) if (e != i0 && p[e] > p[i1]) i1 = e;
    float e0 = __expf(p[i0]), e1 = __expf(p[i1]);
    float wsum = __builtin_amdgcn_rcpf(e0 + e1);
    float w0 = e0 * wsum, w1 = e1 * wsum;
    int pos0 = atomicAdd(&cnt[i0], 1);
    ltok[i0 * NTOK + pos0] = (t << 1);     lw[i0 * NTOK + pos0] = w0;
    int pos1 = atomicAdd(&cnt[i1], 1);
    ltok[i1 * NTOK + pos1] = (t << 1) | 1; lw[i1 * NTOK + pos1] = w1;
  }
}

// ---------------- B-tile staging (32 K x 256 N -> K-major bf16 pairs) ---------

template <bool BF16SRC>
__device__ __forceinline__ void stage_tile(const void* Wsrc, int ldw, int k0, int nb,
                                           unsigned int* dst, int tid, int ktot) {
  unsigned int* d = dst + tid * B_RS;
  if constexpr (BF16SRC) {
    const unsigned short* col = (const unsigned short*)Wsrc + (size_t)k0 * ldw + nb + tid;
    if (k0 + 64 < ktot) __builtin_prefetch(col + (size_t)64 * ldw, 0, 0);
#pragma unroll
    for (int i = 0; i < 16; ++i) {
      unsigned int lo = col[(size_t)(2 * i) * ldw];
      unsigned int hi = col[(size_t)(2 * i + 1) * ldw];
      d[i] = lo | (hi << 16);
    }
  } else {
    const float* col = (const float*)Wsrc + (size_t)k0 * ldw + nb + tid;
    if (k0 + 64 < ktot) __builtin_prefetch(col + (size_t)64 * ldw, 0, 0);
#pragma unroll
    for (int i = 0; i < 16; ++i) {
      float f0 = col[(size_t)(2 * i) * ldw];
      float f1 = col[(size_t)(2 * i + 1) * ldw];
      d[i] = (unsigned int)f2bf(f0) | ((unsigned int)f2bf(f1) << 16);
    }
  }
}

// ---------------- fused expert FFN (gathered grouped GEMM) ----------------

template <bool WB>   // WB: weights pre-converted to bf16 in workspace
__global__ __launch_bounds__(256) void ffn(
    const unsigned int* __restrict__ xb,   // x as bf16 pairs [NTOK][DDIM/2]
    const void* __restrict__ Wg,           // [E][D][F]  (f32 or bf16)
    const void* __restrict__ Wu,           // [E][D][F]
    const void* __restrict__ Wd,           // [E][F][D]
    const int* __restrict__ cnt,
    const int* __restrict__ ltok,
    const float* __restrict__ lw,
    float* __restrict__ out2)              // [NTOK*2][D]
{
  __shared__ unsigned int sA[TILE_M * XA_RS];       // gathered X, bf16 pairs
  __shared__ unsigned int sF[TILE_M * FU_RS];       // fuse = silu(g)*u, bf16 pairs
  __shared__ unsigned int sBg[2][NCOLS * B_RS];     // ping-pong B stages
  __shared__ unsigned int sBu[2][NCOLS * B_RS];
  __shared__ float swt[TILE_M];
  __shared__ int   sslot[TILE_M];

  const int e    = blockIdx.x >> 6;                 // 64 tiles per expert
  const int tile = blockIdx.x & 63;
  const int count = cnt[e];
  if (tile * TILE_M >= count) return;               // uniform early exit

  const int tid   = threadIdx.x;
  const int lane  = tid & 31;
  const int wv    = tid >> 5;
  const int mh    = wv >> 2;                        // 0/1 : 16-row half of M tile
  const int nq    = wv & 3;                         // 0..3 : 64-col quarter of N pass
  const int laneN = lane & 15;
  const int laneHi = lane >> 4;
  const int rowA  = mh * 16 + laneN;                // A-matrix row this lane owns

  if (tid < TILE_M) {
    int idx = tile * TILE_M + tid;
    if (idx < count) { sslot[tid] = ltok[e * NTOK + idx]; swt[tid] = lw[e * NTOK + idx]; }
    else             { sslot[tid] = -1;                   swt[tid] = 0.f; }
  }
  __syncthreads();

  // gather X rows (already bf16) into LDS
  for (int m = 0; m < TILE_M; ++m) {
    int sl = sslot[m];
    int tok = (sl >= 0) ? (sl >> 1) : 0;
    const unsigned int* src = xb + (size_t)tok * (DDIM / 2);
    sA[m * XA_RS + tid]       = src[tid];
    sA[m * XA_RS + 256 + tid] = src[256 + tid];
  }
  __syncthreads();

  const size_t esz = WB ? 2u : 4u;
  const char* wg = (const char*)Wg + (size_t)e * DDIM * FDIM * esz;
  const char* wu = (const char*)Wu + (size_t)e * DDIM * FDIM * esz;
  const char* wd = (const char*)Wd + (size_t)e * FDIM * DDIM * esz;

  // ---- GEMM1: [32 x 1024] @ [1024 x 2048] gating & up, fused SiLU*up ----
  for (int np = 0; np < FDIM / NCOLS; ++np) {
    const int nb = np * NCOLS;
    v8f ag[4], au[4];
#pragma unroll
    for (int t = 0; t < 4; ++t) { ag[t] = (v8f){}; au[t] = (v8f){}; }

    stage_tile<WB>(wg, FDIM, 0, nb, sBg[0], tid, DDIM);
    stage_tile<WB>(wu, FDIM, 0, nb, sBu[0], tid, DDIM);
    __syncthreads();

    for (int i = 0; i < DDIM / 32; ++i) {
      const int cur = i & 1;
      if (i + 1 < DDIM / 32) {            // stage next K-step while computing
        stage_tile<WB>(wg, FDIM, (i + 1) * 32, nb, sBg[cur ^ 1], tid, DDIM);
        stage_tile<WB>(wu, FDIM, (i + 1) * 32, nb, sBu[cur ^ 1], tid, DDIM);
      }
      const int k0 = i * 32;
      union { v16bf v; uint4 q[2]; } a;
      {
        const uint4* pa = (const uint4*)(sA + rowA * XA_RS + (k0 >> 1) + laneHi * 4);
        a.q[0] = pa[0];                   // K {0..7}  ({8..15} for upper lanes)
        a.q[1] = pa[2];                   // K {16..23} ({24..31})
      }
#pragma unroll
      for (int t = 0; t < 4; ++t) {
        int n = nq * 64 + t * 16 + laneN;
        union { v16bf v; uint4 q[2]; } bg, bu;
        const uint4* pg = (const uint4*)(sBg[cur] + n * B_RS + laneHi * 4);
        const uint4* pu = (const uint4*)(sBu[cur] + n * B_RS + laneHi * 4);
        bg.q[0] = pg[0]; bg.q[1] = pg[2];
        bu.q[0] = pu[0]; bu.q[1] = pu[2];
        ag[t] = __builtin_amdgcn_wmma_f32_16x16x32_bf16(false, a.v, false, bg.v,
                                                        (short)0, ag[t], false, false);
        au[t] = __builtin_amdgcn_wmma_f32_16x16x32_bf16(false, a.v, false, bu.v,
                                                        (short)0, au[t], false, false);
      }
      __syncthreads();
    }

    // epilogue: fuse = silu(g) * u, bf16 into LDS
    unsigned short* fu16 = (unsigned short*)sF;
#pragma unroll
    for (int t = 0; t < 4; ++t) {
      int col = nb + nq * 64 + t * 16 + laneN;
#pragma unroll
      for (int r = 0; r < 8; ++r) {
        int mAbs = mh * 16 + r + laneHi * 8;     // C/D layout: VGPR r -> M=r / r+8
        float g = ag[t][r], u = au[t][r];
        float v = g * __builtin_amdgcn_rcpf(1.f + __expf(-g)) * u;
        fu16[mAbs * (FU_RS * 2) + col] = f2bf(v);
      }
    }
  }
  __syncthreads();

  // ---- GEMM2: fuse[32 x 2048] @ Wd[2048 x 1024], scale by routing weight ----
  for (int np = 0; np < DDIM / NCOLS; ++np) {
    const int nb = np * NCOLS;
    v8f ac[4];
#pragma unroll
    for (int t = 0; t < 4; ++t) ac[t] = (v8f){};

    stage_tile<WB>(wd, DDIM, 0, nb, sBg[0], tid, FDIM);
    __syncthreads();

    for (int i = 0; i < FDIM / 32; ++i) {
      const int cur = i & 1;
      if (i + 1 < FDIM / 32)
        stage_tile<WB>(wd, DDIM, (i + 1) * 32, nb, sBg[cur ^ 1], tid, FDIM);
      const int k0 = i * 32;
      union { v16bf v; uint4 q[2]; } a;
      {
        const uint4* pa = (const uint4*)(sF + rowA * FU_RS + (k0 >> 1) + laneHi * 4);
        a.q[0] = pa[0];
        a.q[1] = pa[2];
      }
#pragma unroll
      for (int t = 0; t < 4; ++t) {
        int n = nq * 64 + t * 16 + laneN;
        union { v16bf v; uint4 q[2]; } bd;
        const uint4* pd = (const uint4*)(sBg[cur] + n * B_RS + laneHi * 4);
        bd.q[0] = pd[0]; bd.q[1] = pd[2];
        ac[t] = __builtin_amdgcn_wmma_f32_16x16x32_bf16(false, a.v, false, bd.v,
                                                        (short)0, ac[t], false, false);
      }
      __syncthreads();
    }

    // weighted scatter-store into per-(token,k) slot -> deterministic
#pragma unroll
    for (int t = 0; t < 4; ++t) {
      int col = nb + nq * 64 + t * 16 + laneN;
#pragma unroll
      for (int r = 0; r < 8; ++r) {
        int mAbs = mh * 16 + r + laneHi * 8;
        int sl = sslot[mAbs];
        if (sl >= 0) out2[(size_t)sl * DDIM + col] = ac[t][r] * swt[mAbs];
      }
    }
  }
}

__global__ void reduce2(const float* __restrict__ out2, float* __restrict__ out) {
  int i = blockIdx.x * blockDim.x + threadIdx.x;
  if (i < NTOK * DDIM) {
    int t = i >> 10, c = i & 1023;
    out[i] = out2[(size_t)(2 * t) * DDIM + c] + out2[(size_t)(2 * t + 1) * DDIM + c];
  }
}

// ---------------- host entry ----------------

extern "C" void kernel_launch(void* const* d_in, const int* in_sizes, int n_in,
                              void* d_out, int out_size, void* d_ws, size_t ws_size,
                              hipStream_t stream) {
  const float* x  = (const float*)d_in[0];
  const float* R  = (const float*)d_in[1];
  const float* Wg = (const float*)d_in[2];
  const float* Wu = (const float*)d_in[3];
  const float* Wd = (const float*)d_in[4];
  float* out = (float*)d_out;

  char* ws = (char*)d_ws;
  const size_t MB = 1024u * 1024u;
  unsigned int* xb = (unsigned int*)ws;                      // 4 MB
  int*   cnt  = (int*)(ws + 4 * MB);                         // 32 B
  int*   ltok = (int*)(ws + 4 * MB + 256);                   // 64 KB
  float* lw   = (float*)(ws + 4 * MB + 256 + NEXP * NTOK * 4);
  float* out2 = (float*)(ws + 4 * MB + 256 + 2 * NEXP * NTOK * 4);   // 16 MB
  unsigned int* wgbf = (unsigned int*)(ws + 32 * MB);        // 32 MB each
  unsigned int* wubf = (unsigned int*)(ws + 64 * MB);
  unsigned int* wdbf = (unsigned int*)(ws + 96 * MB);
  const bool preconv = ws_size >= 128 * MB;

  const int WPAIRS = NEXP * DDIM * FDIM / 2;                 // 8,388,608 per tensor

  zero_cnt<<<1, 32, 0, stream>>>(cnt);
  cvt_x<<<(NTOK * DDIM / 2 + 255) / 256, 256, 0, stream>>>(x, xb);
  router<<<NTOK / 8, 256, 0, stream>>>(x, R, cnt, ltok, lw);

  if (preconv) {
    cvt_w<<<(WPAIRS + 255) / 256, 256, 0, stream>>>(Wg, wgbf, WPAIRS);
    cvt_w<<<(WPAIRS + 255) / 256, 256, 0, stream>>>(Wu, wubf, WPAIRS);
    cvt_w<<<(WPAIRS + 255) / 256, 256, 0, stream>>>(Wd, wdbf, WPAIRS);
    ffn<true><<<NEXP * (NTOK / TILE_M), 256, 0, stream>>>(
        xb, wgbf, wubf, wdbf, cnt, ltok, lw, out2);
  } else {
    ffn<false><<<NEXP * (NTOK / TILE_M), 256, 0, stream>>>(
        xb, Wg, Wu, Wd, cnt, ltok, lw, out2);
  }
  reduce2<<<(NTOK * DDIM + 255) / 256, 256, 0, stream>>>(out2, out);
}
